// RWKV6_CrossAttBlock_78391743086616
// MI455X (gfx1250) — matrix operand order repaired and checked by reading, not compiled, not run on hardware
//
#include <hip/hip_runtime.h>

#define HD     512
#define NHEADS 8
#define DHEAD  64
#define RLOW   32
#define GLRK   64
#define IDIM   1792
#define BBATCH 4
#define TSEQ   2048
#define MM     (BBATCH * TSEQ)
#define EPSV   1e-5f

typedef unsigned short u16;
typedef __attribute__((ext_vector_type(16))) __bf16 v16bf;
typedef __attribute__((ext_vector_type(8)))  float  v8f;

struct U4x2 { uint4 lo; uint4 hi; };
union FragU { U4x2 p; v16bf v; };

__device__ __forceinline__ u16 f2b(float f) {
  unsigned u = __float_as_uint(f);
  return (u16)((u + 0x7fffu + ((u >> 16) & 1u)) >> 16);
}

// ---------------------------------------------------------------------------
// WMMA GEMM: C[M,N] = act(A[M,K] * Bt[N,K]^T + bias)
// A: bf16 row-major (lda), Bt: bf16 transposed weights [N][K], C: fp32,
// Cb: optional bf16 mirror.
// Block tile 64(M) x 128(N), 128 threads = 4 waves; each wave 32x64 patch
// (2x4 of 16x16), K stepped by 32 via v_wmma_f32_16x16x32_bf16.
// Global->LDS staging uses CDNA5 async-to-LDS DMA (ASYNCcnt).
// ACT: 0 none, 1 tanh, 2 sqrelu, 3 sigmoid, 4 exp(-exp(x))
// ---------------------------------------------------------------------------
template<int ACT, bool HAS_BIAS, bool WF32, bool WBF16>
__global__ __launch_bounds__(128)
void gemm_wmma_bf16(const u16* __restrict__ A, int lda,
                    const u16* __restrict__ Bt,
                    float* __restrict__ C, u16* __restrict__ Cb,
                    const float* __restrict__ bias,
                    int N, int K) {
  __shared__ u16 lA[64 * 40];    // 64 rows (M), 32 K + pad
  __shared__ u16 lB[128 * 40];   // 128 rows (N), 32 K + pad
  const int n0   = blockIdx.x * 128;
  const int m0   = blockIdx.y * 64;
  const int tid  = threadIdx.x;
  const int lane = tid & 31;
  const int wave = tid >> 5;
  const int wm   = (wave >> 1) * 32;
  const int wn   = (wave & 1) * 64;
  const int hrow = lane & 15;
  const int hi   = lane >> 4;            // 0 or 1
  const int row  = tid >> 1;             // 0..63
  const int half = (tid & 1) * 16;       // 0 or 16

  // Zero-fill out-of-range B rows once; async copies never touch them,
  // so they stay zero across the whole K loop.
  #pragma unroll
  for (int rr0 = 0; rr0 < 128; rr0 += 64) {
    int rr = rr0 + row;
    if (n0 + rr >= N) {
      uint4 z = {0u, 0u, 0u, 0u};
      uint4* dz = (uint4*)(lB + rr * 40 + half);
      dz[0] = z; dz[1] = z;
    }
  }

  v8f acc[2][4];
  #pragma unroll
  for (int i = 0; i < 2; i++)
    #pragma unroll
    for (int j = 0; j < 4; j++)
      #pragma unroll
      for (int e = 0; e < 8; e++) acc[i][j][e] = 0.f;

  const unsigned ldsA  = (unsigned)(uintptr_t)(lA + row * 40 + half);
  const unsigned ldsB0 = (unsigned)(uintptr_t)(lB + row * 40 + half);
  const unsigned ldsB1 = (unsigned)(uintptr_t)(lB + (row + 64) * 40 + half);
  const int nr0 = n0 + row, nr1 = n0 + row + 64;

  for (int k0 = 0; k0 < K; k0 += 32) {
    // async DMA: global -> LDS, 16B per lane, tracked by ASYNCcnt
    {
      const u16* ga = A + (size_t)(m0 + row) * lda + k0 + half;
      asm volatile("global_load_async_to_lds_b128 %0, %1, off"
                   :: "v"(ldsA), "v"(ga) : "memory");
    }
    if (nr0 < N) {
      const u16* gb = Bt + (size_t)nr0 * K + k0 + half;
      asm volatile("global_load_async_to_lds_b128 %0, %1, off"
                   :: "v"(ldsB0), "v"(gb) : "memory");
    }
    if (nr1 < N) {
      const u16* gb = Bt + (size_t)nr1 * K + k0 + half;
      asm volatile("global_load_async_to_lds_b128 %0, %1, off"
                   :: "v"(ldsB1), "v"(gb) : "memory");
    }
    if (k0 + 32 < K)
      __builtin_prefetch(A + (size_t)(m0 + row) * lda + k0 + 32 + half, 0, 1);
    asm volatile("s_wait_asynccnt 0x0" ::: "memory");
    __syncthreads();

    const int abase = hi ? 8 : 0;    // A lane K-base (chunks at abase, abase+16)
    const int bbase = hi ? 8 : 0;    // B lane K-base (chunks at bbase, bbase+16)
    FragU fa[2], fb[4];
    #pragma unroll
    for (int i = 0; i < 2; i++) {
      int r = wm + i * 16 + hrow;
      fa[i].p.lo = *(const uint4*)(lA + r * 40 + abase);
      fa[i].p.hi = *(const uint4*)(lA + r * 40 + abase + 16);
    }
    #pragma unroll
    for (int j = 0; j < 4; j++) {
      int r = wn + j * 16 + hrow;
      fb[j].p.lo = *(const uint4*)(lB + r * 40 + bbase);
      fb[j].p.hi = *(const uint4*)(lB + r * 40 + bbase + 16);
    }
    #pragma unroll
    for (int i = 0; i < 2; i++)
      #pragma unroll
      for (int j = 0; j < 4; j++)
        acc[i][j] = __builtin_amdgcn_wmma_f32_16x16x32_bf16(
            false, fa[i].v, false, fb[j].v, (short)0, acc[i][j], false, false);
    __syncthreads();
  }

  #pragma unroll
  for (int j = 0; j < 4; j++) {
    int col = n0 + wn + j * 16 + hrow;
    if (col >= N) continue;
    float bv = HAS_BIAS ? bias[col] : 0.f;
    #pragma unroll
    for (int i = 0; i < 2; i++) {
      int rbase = m0 + wm + i * 16 + hi * 8;
      #pragma unroll
      for (int e = 0; e < 8; e++) {
        float x = acc[i][j][e] + bv;
        if (ACT == 1)      x = tanhf(x);
        else if (ACT == 2) { x = fmaxf(x, 0.f); x = x * x; }
        else if (ACT == 3) x = 1.f / (1.f + expf(-x));
        else if (ACT == 4) x = expf(-expf(x));
        size_t o = (size_t)(rbase + e) * N + col;
        if (WF32)  C[o]  = x;
        if (WBF16) Cb[o] = f2b(x);
      }
    }
  }
}

// ---------------------------------------------------------------------------
// LayerNorm over H=512, one token per 256-thread block (2 elems/thread)
// ---------------------------------------------------------------------------
__global__ __launch_bounds__(256)
void ln_kernel(const float* __restrict__ x, const float* __restrict__ g,
               const float* __restrict__ b, float* __restrict__ y) {
  __shared__ float red[256];
  const int rowi = blockIdx.x, t = threadIdx.x;
  const float* xr = x + (size_t)rowi * HD;
  float a0 = xr[t], a1 = xr[t + 256];
  red[t] = a0 + a1; __syncthreads();
  for (int s = 128; s > 0; s >>= 1) { if (t < s) red[t] += red[t + s]; __syncthreads(); }
  float mu = red[0] * (1.f / HD); __syncthreads();
  float d0 = a0 - mu, d1 = a1 - mu;
  red[t] = d0 * d0 + d1 * d1; __syncthreads();
  for (int s = 128; s > 0; s >>= 1) { if (t < s) red[t] += red[t + s]; __syncthreads(); }
  float rs = rsqrtf(red[0] * (1.f / HD) + EPSV);
  float* yr = y + (size_t)rowi * HD;
  yr[t]       = d0 * rs * g[t]       + b[t];
  yr[t + 256] = d1 * rs * g[t + 256] + b[t + 256];
}

// ---------------------------------------------------------------------------
// token-shift lerp mix -> bf16:  out = x + (shift(x) - x) * coef
// VEC: coef is a per-channel [H] vector; else full [M,H] buffer
// ---------------------------------------------------------------------------
template<bool VEC>
__global__ __launch_bounds__(256)
void mix_kernel(const float* __restrict__ base, const float* __restrict__ coef,
                u16* __restrict__ out, int total) {
  int idx = blockIdx.x * blockDim.x + threadIdx.x;
  if (idx >= total) return;
  int h = idx & (HD - 1);
  int t = (idx / HD) & (TSEQ - 1);
  float xc = base[idx];
  float xp = t ? base[idx - HD] : 0.f;
  float m  = VEC ? coef[h] : coef[idx];
  out[idx] = f2b(xc + (xp - xc) * m);
}

// ---------------------------------------------------------------------------
// RWKV6 recurrence: one block per (batch, head), 64 threads (Dv lanes),
// state S[Dk=64] per thread in VGPRs, r/k/decay/u staged via LDS.
// ---------------------------------------------------------------------------
__global__ __launch_bounds__(64)
void rwkv_scan_kernel(const float* __restrict__ r, const float* __restrict__ k,
                      const float* __restrict__ v, const float* __restrict__ w,
                      const float* __restrict__ u, float* __restrict__ o) {
  const int bh = blockIdx.x;
  const int b  = bh / NHEADS, h = bh % NHEADS;
  const int j  = threadIdx.x;
  __shared__ float sr[DHEAD], sk[DHEAD], sw[DHEAD], su[DHEAD];
  su[j] = u[h * DHEAD + j];
  float S[DHEAD];
  #pragma unroll
  for (int i = 0; i < DHEAD; i++) S[i] = 0.f;
  const size_t base = (size_t)b * TSEQ * HD + h * DHEAD + j;
  for (int t = 0; t < TSEQ; t++) {
    size_t idx = base + (size_t)t * HD;
    sr[j] = r[idx]; sk[j] = k[idx]; sw[j] = w[idx];
    float vj = v[idx];
    __syncthreads();
    float acc = 0.f;
    #pragma unroll
    for (int i = 0; i < DHEAD; i++) {
      float kv = sk[i] * vj;
      acc  = fmaf(sr[i], fmaf(su[i], kv, S[i]), acc);
      S[i] = fmaf(sw[i], S[i], kv);
    }
    o[idx] = acc;
    __syncthreads();
  }
}

// ---------------------------------------------------------------------------
// GroupNorm(HEADS groups) + silu-gate, fp32 -> bf16 (Wo GEMM operand)
// ---------------------------------------------------------------------------
__global__ __launch_bounds__(512)
void gn_gate_kernel(const float* __restrict__ o, const float* __restrict__ g,
                    const float* __restrict__ gg, const float* __restrict__ gb,
                    u16* __restrict__ out) {
  __shared__ float red[512];
  const int rowi = blockIdx.x, c = threadIdx.x;
  const int gbase = c & ~63;
  float x = o[(size_t)rowi * HD + c];
  red[c] = x; __syncthreads();
  for (int s = 32; s > 0; s >>= 1) { if ((c & 63) < s) red[c] += red[c + s]; __syncthreads(); }
  float mu = red[gbase] * (1.f / DHEAD); __syncthreads();
  float d = x - mu;
  red[c] = d * d; __syncthreads();
  for (int s = 32; s > 0; s >>= 1) { if ((c & 63) < s) red[c] += red[c + s]; __syncthreads(); }
  float y = d * rsqrtf(red[gbase] * (1.f / DHEAD) + EPSV) * gg[c] + gb[c];
  float gv = g[(size_t)rowi * HD + c];
  y *= gv / (1.f + expf(-gv));
  out[(size_t)rowi * HD + c] = f2b(y);
}

__global__ __launch_bounds__(256)
void add_inplace_kernel(float* __restrict__ dst, const float* __restrict__ a, int n) {
  int i = blockIdx.x * blockDim.x + threadIdx.x;
  if (i < n) dst[i] += a[i];
}
__global__ __launch_bounds__(256)
void fma_inplace_kernel(float* __restrict__ dst, const float* __restrict__ a,
                        const float* __restrict__ b, int n) {
  int i = blockIdx.x * blockDim.x + threadIdx.x;
  if (i < n) dst[i] = fmaf(a[i], b[i], dst[i]);
}

// weights: fp32 [K,N] -> bf16 transposed [N,K]
__global__ __launch_bounds__(256)
void wt_transpose_kernel(const float* __restrict__ W, u16* __restrict__ Wt, int K, int N) {
  int idx = blockIdx.x * blockDim.x + threadIdx.x;
  if (idx >= K * N) return;
  int kk = idx / N, n = idx - kk * N;
  Wt[(size_t)n * K + kk] = f2b(W[idx]);
}
// Wx2 [H,5,R] -> per-n bf16 [H][R] blocks: Wt[n][h][r] = W[h][n][r]
__global__ __launch_bounds__(256)
void wx2_conv_kernel(const float* __restrict__ W, u16* __restrict__ Wt) {
  int idx = blockIdx.x * blockDim.x + threadIdx.x;
  if (idx >= HD * 5 * RLOW) return;
  int h = idx / (5 * RLOW);
  int n = (idx / RLOW) % 5;
  int rr = idx % RLOW;
  Wt[((size_t)n * HD + h) * RLOW + rr] = f2b(W[idx]);
}

// ---------------------------------------------------------------------------
// Host side
// ---------------------------------------------------------------------------
static inline int cdiv(int a, int b) { return (a + b - 1) / b; }
#define GGRID(N_) dim3(cdiv((N_), 128), MM / 64)

struct AttnW {
  const float *mu_x, *x_bias, *bw, *u, *gn_g, *gn_b;
  u16 *x1t, *x2t, *rt, *kt, *vt, *gt, *w1t, *w2t, *ot;
};
struct Scr {
  float *S_MUS, *S_R, *S_K, *S_V, *S_G, *S_W, *S_O;
  u16 *B_A, *B_MIX, *B_X1, *B_WMID;
};

static void run_attn(const float* qsrc, const float* kvsrc, const AttnW& w,
                     const Scr& s, float* outp, hipStream_t stream) {
  const int MH = MM * HD;
  dim3 blk(128);
  // xin = kv + dkv*mu_x (bf16)
  mix_kernel<true><<<cdiv(MH, 256), 256, 0, stream>>>(kvsrc, w.mu_x, s.B_A, MH);
  // X1 = tanh(xin @ Wx1) -> bf16 [M,160]
  gemm_wmma_bf16<1, false, false, true><<<GGRID(5 * RLOW), blk, 0, stream>>>(
      s.B_A, HD, w.x1t, nullptr, s.B_X1, nullptr, 5 * RLOW, HD);
  for (int n = 0; n < 5; n++) {
    // mus_n = X1_n @ Wx2_n^T + x_bias_n  -> fp32 [M,H]
    gemm_wmma_bf16<0, true, true, false><<<GGRID(HD), blk, 0, stream>>>(
        s.B_X1 + n * RLOW, 5 * RLOW, w.x2t + (size_t)n * HD * RLOW,
        s.S_MUS, nullptr, w.x_bias + n * HD, HD, RLOW);
    const float* basep = (n == 0 || n == 4) ? qsrc : kvsrc;  // r,g from q; w,k,v from kv
    mix_kernel<false><<<cdiv(MH, 256), 256, 0, stream>>>(basep, s.S_MUS, s.B_MIX, MH);
    switch (n) {
      case 0:
        gemm_wmma_bf16<0, false, true, false><<<GGRID(HD), blk, 0, stream>>>(
            s.B_MIX, HD, w.rt, s.S_R, nullptr, nullptr, HD, HD);
        break;
      case 1:
        gemm_wmma_bf16<1, false, false, true><<<GGRID(GLRK), blk, 0, stream>>>(
            s.B_MIX, HD, w.w1t, nullptr, s.B_WMID, nullptr, GLRK, HD);
        gemm_wmma_bf16<4, true, true, false><<<GGRID(HD), blk, 0, stream>>>(
            s.B_WMID, GLRK, w.w2t, s.S_W, nullptr, w.bw, HD, GLRK);  // decay = exp(-exp(w+bw))
        break;
      case 2:
        gemm_wmma_bf16<0, false, true, false><<<GGRID(HD), blk, 0, stream>>>(
            s.B_MIX, HD, w.kt, s.S_K, nullptr, nullptr, HD, HD);
        break;
      case 3:
        gemm_wmma_bf16<0, false, true, false><<<GGRID(HD), blk, 0, stream>>>(
            s.B_MIX, HD, w.vt, s.S_V, nullptr, nullptr, HD, HD);
        break;
      case 4:
        gemm_wmma_bf16<0, false, true, false><<<GGRID(HD), blk, 0, stream>>>(
            s.B_MIX, HD, w.gt, s.S_G, nullptr, nullptr, HD, HD);
        break;
    }
  }
  rwkv_scan_kernel<<<BBATCH * NHEADS, 64, 0, stream>>>(s.S_R, s.S_K, s.S_V, s.S_W, w.u, s.S_O);
  gn_gate_kernel<<<MM, 512, 0, stream>>>(s.S_O, s.S_G, w.gn_g, w.gn_b, s.B_MIX);
  gemm_wmma_bf16<0, false, true, false><<<GGRID(HD), blk, 0, stream>>>(
      s.B_MIX, HD, w.ot, outp, nullptr, nullptr, HD, HD);
}

extern "C" void kernel_launch(void* const* d_in, const int* in_sizes, int n_in,
                              void* d_out, int out_size, void* d_ws, size_t ws_size,
                              hipStream_t stream) {
  (void)in_sizes; (void)n_in; (void)out_size; (void)ws_size;
  const float* query  = (const float*)d_in[0];
  const float* keyval = (const float*)d_in[1];
  const float* pre_g = (const float*)d_in[2];
  const float* pre_b = (const float*)d_in[3];
  const float* qn_g  = (const float*)d_in[4];
  const float* qn_b  = (const float*)d_in[5];
  const float* kn_g  = (const float*)d_in[6];
  const float* kn_b  = (const float*)d_in[7];
  const float* fn_g  = (const float*)d_in[8];
  const float* fn_b  = (const float*)d_in[9];
  // attn params: mu_x,Wx1,Wx2,x_bias,Wr,Wk,Wv,Wg,Ww1,Ww2,bw,u,gn_g,gn_b,Wo
  const int I_SA = 10, I_CA = 25, I_FFN = 40;
  auto P = [&](int base, int i) { return (const float*)d_in[base + i]; };

  char* ws = (char*)d_ws;
  size_t off = 0;
  auto alloc = [&](size_t bytes) -> char* {
    char* p = ws + off;
    off += (bytes + 255) & ~(size_t)255;
    return p;
  };
  const size_t MHf = (size_t)MM * HD * sizeof(float);
  const size_t MHb = (size_t)MM * HD * sizeof(u16);
  float* S_Q   = (float*)alloc(MHf);
  float* S_KV  = (float*)alloc(MHf);
  Scr s;
  s.S_MUS = (float*)alloc(MHf);
  s.S_R   = (float*)alloc(MHf);
  s.S_K   = (float*)alloc(MHf);
  s.S_V   = (float*)alloc(MHf);
  s.S_G   = (float*)alloc(MHf);
  s.S_W   = (float*)alloc(MHf);
  s.S_O   = (float*)alloc(MHf);
  s.B_A    = (u16*)alloc(MHb);
  s.B_MIX  = (u16*)alloc(MHb);
  s.B_X1   = (u16*)alloc((size_t)MM * 5 * RLOW * sizeof(u16));
  s.B_WMID = (u16*)alloc((size_t)MM * GLRK * sizeof(u16));
  u16* B_KEY = (u16*)alloc((size_t)MM * IDIM * sizeof(u16));

  // ---- convert weights to transposed bf16 ----
  auto tconv = [&](const float* W, int K, int N) -> u16* {
    u16* Wt = (u16*)alloc((size_t)K * N * sizeof(u16));
    wt_transpose_kernel<<<cdiv(K * N, 256), 256, 0, stream>>>(W, Wt, K, N);
    return Wt;
  };
  AttnW aw[2];
  for (int a = 0; a < 2; a++) {
    int base = a ? I_CA : I_SA;
    AttnW& w = aw[a];
    w.mu_x  = P(base, 0);
    w.x_bias= P(base, 3);
    w.bw    = P(base, 10);
    w.u     = P(base, 11);
    w.gn_g  = P(base, 12);
    w.gn_b  = P(base, 13);
    w.x1t = tconv(P(base, 1), HD, 5 * RLOW);
    w.x2t = (u16*)alloc((size_t)5 * HD * RLOW * sizeof(u16));
    wx2_conv_kernel<<<cdiv(HD * 5 * RLOW, 256), 256, 0, stream>>>(P(base, 2), w.x2t);
    w.rt  = tconv(P(base, 4), HD, HD);
    w.kt  = tconv(P(base, 5), HD, HD);
    w.vt  = tconv(P(base, 6), HD, HD);
    w.gt  = tconv(P(base, 7), HD, HD);
    w.w1t = tconv(P(base, 8), HD, GLRK);
    w.w2t = tconv(P(base, 9), GLRK, HD);
    w.ot  = tconv(P(base, 14), HD, HD);
  }
  u16* f_kt = tconv(P(I_FFN, 1), HD, IDIM);
  u16* f_vt = tconv(P(I_FFN, 2), IDIM, HD);
  u16* f_rt = tconv(P(I_FFN, 4), HD, HD);
  const float* f_mu_k = P(I_FFN, 0);
  const float* f_mu_r = P(I_FFN, 3);

  float* residual = (float*)d_out;
  const int MH = MM * HD;

  // residual = LN(query, pre)
  ln_kernel<<<MM, 256, 0, stream>>>(query, pre_g, pre_b, residual);
  // q = LN(residual, qn);  kv = LN(keyval, kn)
  ln_kernel<<<MM, 256, 0, stream>>>(residual, qn_g, qn_b, S_Q);
  ln_kernel<<<MM, 256, 0, stream>>>(keyval, kn_g, kn_b, S_KV);
  // self-attn then cross-attn (r/g from q, k/v/w from kv)
  run_attn(S_Q, S_Q, aw[0], s, S_Q, stream);
  run_attn(S_Q, S_KV, aw[1], s, S_Q, stream);
  // residual += q
  add_inplace_kernel<<<cdiv(MH, 256), 256, 0, stream>>>(residual, S_Q, MH);
  // q = LN(residual, fn)
  ln_kernel<<<MM, 256, 0, stream>>>(residual, fn_g, fn_b, S_Q);
  // FFN: key = sqrelu((x + d*mu_k) @ Wk); val = key @ Wv; rec = sigmoid((x + d*mu_r) @ Wr)
  mix_kernel<true><<<cdiv(MH, 256), 256, 0, stream>>>(S_Q, f_mu_k, s.B_MIX, MH);
  gemm_wmma_bf16<2, false, false, true><<<GGRID(IDIM), 128, 0, stream>>>(
      s.B_MIX, HD, f_kt, nullptr, B_KEY, nullptr, IDIM, HD);
  gemm_wmma_bf16<0, false, true, false><<<GGRID(HD), 128, 0, stream>>>(
      B_KEY, IDIM, f_vt, s.S_V, nullptr, nullptr, HD, IDIM);
  mix_kernel<true><<<cdiv(MH, 256), 256, 0, stream>>>(S_Q, f_mu_r, s.B_MIX, MH);
  gemm_wmma_bf16<3, false, true, false><<<GGRID(HD), 128, 0, stream>>>(
      s.B_MIX, HD, f_rt, s.S_G, nullptr, nullptr, HD, HD);
  // out = residual + rec * val
  fma_inplace_kernel<<<cdiv(MH, 256), 256, 0, stream>>>(residual, s.S_G, s.S_V, MH);
}
